// AutoregressiveDecoder_86457691668934
// MI455X (gfx1250) — matrix operand order repaired and checked by reading, not compiled, not run on hardware
//
#include <hip/hip_runtime.h>
#include <hip/hip_bf16.h>

typedef __attribute__((ext_vector_type(16))) __bf16 v16bf;
typedef __attribute__((ext_vector_type(8)))  __bf16 v8bf;
typedef __attribute__((ext_vector_type(8)))  float  v8f;

// ---------- helpers ----------

__device__ __forceinline__ __bf16 f2bf(float f) {
  union { float f; unsigned u; } x; x.f = f;
  unsigned r = x.u + 0x7FFFu + ((x.u >> 16) & 1u);   // round-to-nearest-even
  unsigned short h = (unsigned short)(r >> 16);
  __bf16 o; __builtin_memcpy(&o, &h, sizeof(o));
  return o;
}

__device__ __forceinline__ float sigm(float x) { return 1.0f / (1.0f + __expf(-x)); }

// 16-bit A-matrix 16x32 fragment per ISA layout:
// lane L: m = L%16; kbase = (L>=16)?8:0; elems 0..7 = K[k0+kbase+0..7],
// elems 8..15 = K[k0+kbase+16..23]. Two aligned 16B LDS reads.
__device__ __forceinline__ v16bf load_a_frag(const __bf16* row, int k0, int lane) {
  int base = k0 + ((lane & 16) ? 8 : 0);
  v8bf lo = *(const v8bf*)(row + base);
  v8bf hi = *(const v8bf*)(row + base + 16);
  v16bf a;
#pragma unroll
  for (int i = 0; i < 8; ++i) { a[i] = lo[i]; a[i + 8] = hi[i]; }
  return a;
}

__device__ __forceinline__ v8f wmma_bf16(v16bf a, v16bf b, v8f c) {
  return __builtin_amdgcn_wmma_f32_16x16x32_bf16(false, a, false, b, (short)0, c, false, false);
}

// ---------- pre-pass: f32 weights -> bf16 in WMMA-B fragment order ----------
// Packed layout: tile = n_t*(K/32)+kt; elem index = tile*512 + lane*16 + i,
// holding B[k0 + (lane>=16)*8 + (i<8 ? i : i+8)][n0 + lane%16]  (16 halfs/lane).
__global__ void pack_wmma_b(const float* __restrict__ W, unsigned short* __restrict__ dst,
                            int K, int N) {
  int idx = blockIdx.x * 256 + threadIdx.x;
  if (idx >= K * N) return;
  int tile = idx >> 9, within = idx & 511;
  int lane = within >> 4, i = within & 15;
  int ktiles = K >> 5;
  int k0 = (tile % ktiles) << 5;
  int n0 = (tile / ktiles) << 4;
  int k = k0 + ((lane & 16) ? 8 : 0) + (i < 8 ? i : i + 8);
  int n = n0 + (lane & 15);
  union { __bf16 b; unsigned short u; } cv; cv.b = f2bf(W[k * N + n]);
  dst[idx] = cv.u;
}

// ---------- fused autoregressive decoder ----------
// One workgroup = 16 batch rows, 8 waves, runs all 128 steps with LDS state.
__global__ __launch_bounds__(256)
void ar_decoder_fused(const float* __restrict__ z, const float* __restrict__ bos,
                      const float* __restrict__ emb_type, const float* __restrict__ emb_color,
                      const float* __restrict__ W_pos_in, const float* __restrict__ b_pos_in,
                      const float* __restrict__ b_lstm, const float* __restrict__ bh0,
                      const float* __restrict__ bc0, const float* __restrict__ b_len,
                      const float* __restrict__ b_type, const float* __restrict__ W_pos,
                      const float* __restrict__ b_pos, const float* __restrict__ b_color,
                      const unsigned short* __restrict__ Wxb_u, const unsigned short* __restrict__ Whb_u,
                      const unsigned short* __restrict__ Wcolb_u, const unsigned short* __restrict__ Wtypb_u,
                      const unsigned short* __restrict__ Wh0b_u, const unsigned short* __restrict__ Wc0b_u,
                      const unsigned short* __restrict__ Wlenb_u, float* __restrict__ out) {
  const __bf16* Wxb   = (const __bf16*)Wxb_u;
  const __bf16* Whb   = (const __bf16*)Whb_u;
  const __bf16* Wcolb = (const __bf16*)Wcolb_u;
  const __bf16* Wtypb = (const __bf16*)Wtypb_u;
  const __bf16* Wh0b  = (const __bf16*)Wh0b_u;
  const __bf16* Wc0b  = (const __bf16*)Wc0b_u;
  const __bf16* Wlenb = (const __bf16*)Wlenb_u;

  __shared__ __align__(16) float  s_gates[16 * 1024];   // reused as color-logits buffer
  __shared__ __align__(16) float  s_h[16 * 256];
  __shared__ __align__(16) float  s_c[16 * 256];
  __shared__ __align__(16) __bf16 s_hb[16 * 256];
  __shared__ __align__(16) __bf16 s_eb[16 * 256];       // also holds z (bf16) during init
  __shared__ __align__(16) float  s_typ[16 * 32];
  __shared__ float s_pos[16 * 4];
  __shared__ int   s_tidx[16];
  __shared__ int   s_cidx[16 * 3];
  __shared__ float s_redv[192];                         // color argmax partials
  __shared__ int   s_redi[192];
  __shared__ float s_posp[256];                         // pos dot partials

  const int tid  = threadIdx.x;
  const int lane = tid & 31;
  // wave index as a compiler-known scalar: keeps branches on it in SALU and
  // EXEC fully set around every WMMA (ISA: WMMA requires EXEC == all 1s).
  const int wv   = __builtin_amdgcn_readfirstlane(tid >> 5);
  const int am   = lane & 15;                  // A-fragment row
  const int dn   = lane & 15;                  // D column
  const int dm0  = (lane & 16) ? 8 : 0;        // D row base
  const int b0   = blockIdx.x * 16;

  const long long OFF_TYPE  = 65536;           // 512*128
  const long long OFF_POS   = 2162688;         // + 512*128*32
  const long long OFF_COLOR = 2424832;         // + 512*128*4

  // ---- init 0: z -> bf16 in s_eb
  for (int idx = tid; idx < 16 * 256; idx += 256) {
    int m = idx >> 8, n = idx & 255;
    s_eb[idx] = f2bf(z[(long long)(b0 + m) * 256 + n]);
  }
  __syncthreads();

  // ---- init 1: h0raw / c0raw / length head via WMMA (40 tile-jobs over 8 waves)
  {
    v16bf az[8];
#pragma unroll
    for (int kt = 0; kt < 8; ++kt) az[kt] = load_a_frag(&s_eb[am * 256], kt * 32, lane);
#pragma unroll 1
    for (int it = 0; it < 5; ++it) {
      int j = wv + it * 8;                      // scalar job id -> scalar branches
      const __bf16* Bw; int n_t, col;
      if (j < 16)      { Bw = Wh0b;  n_t = j;      col = n_t * 16; }
      else if (j < 32) { Bw = Wc0b;  n_t = j - 16; col = 256 + n_t * 16; }
      else             { Bw = Wlenb; n_t = j - 32; col = 512 + n_t * 16; }
      const __bf16* Bt = Bw + (long long)n_t * 4096;   // scalar base (SGPR pair)
      v8f acc = {};
#pragma unroll
      for (int kt = 0; kt < 8; ++kt) {
        v16bf b = *(const v16bf*)(Bt + (kt * 32 + lane) * 16);
        acc = wmma_bf16(az[kt], b, acc);
      }
#pragma unroll
      for (int r = 0; r < 8; ++r) s_gates[(dm0 + r) * 1024 + col + dn] = acc[r];
    }
  }
  __syncthreads();

  // ---- init 2: h0 = tanh(.), c0 = tanh(.), e0 = bos, out_length
  for (int idx = tid; idx < 16 * 256; idx += 256) {
    int m = idx >> 8, n = idx & 255;
    float h = tanhf(s_gates[m * 1024 + n] + bh0[n]);
    float c = tanhf(s_gates[m * 1024 + 256 + n] + bc0[n]);
    s_h[idx] = h; s_hb[idx] = f2bf(h); s_c[idx] = c;
    s_eb[idx] = f2bf(bos[n]);
  }
  for (int idx = tid; idx < 16 * 128; idx += 256) {
    int m = idx >> 7, n = idx & 127;
    out[(long long)(b0 + m) * 128 + n] = s_gates[m * 1024 + 512 + n] + b_len[n];
  }
  __syncthreads();

  // ---- autoregressive loop ----
  for (int t = 0; t < 128; ++t) {
    // stage 1: gates = e@Wx + h@Wh  (64 N-tiles, 8 per wave, K=256)
    {
      v16bf ae[8], ah[8];
#pragma unroll
      for (int kt = 0; kt < 8; ++kt) {
        ae[kt] = load_a_frag(&s_eb[am * 256], kt * 32, lane);
        ah[kt] = load_a_frag(&s_hb[am * 256], kt * 32, lane);
      }
#pragma unroll 1
      for (int j = 0; j < 8; ++j) {
        int n_t = wv * 8 + j;                            // scalar
        const __bf16* Bx = Wxb + (long long)n_t * 4096;  // SGPR bases
        const __bf16* Bh = Whb + (long long)n_t * 4096;
        v8f acc = {};
#pragma unroll
        for (int kt = 0; kt < 8; ++kt) {
          v16bf bx = *(const v16bf*)(Bx + (kt * 32 + lane) * 16);
          acc = wmma_bf16(ae[kt], bx, acc);
          v16bf bh = *(const v16bf*)(Bh + (kt * 32 + lane) * 16);
          acc = wmma_bf16(ah[kt], bh, acc);
        }
#pragma unroll
        for (int r = 0; r < 8; ++r) s_gates[(dm0 + r) * 1024 + n_t * 16 + dn] = acc[r];
      }
    }
    __syncthreads();

    // stage 2: LSTM cell elementwise
    for (int idx = tid; idx < 16 * 256; idx += 256) {
      int m = idx >> 8, n = idx & 255;
      float gi = s_gates[m * 1024 + n]       + b_lstm[n];
      float gf = s_gates[m * 1024 + 256 + n] + b_lstm[256 + n];
      float gg = s_gates[m * 1024 + 512 + n] + b_lstm[512 + n];
      float go = s_gates[m * 1024 + 768 + n] + b_lstm[768 + n];
      float c  = sigm(gf) * s_c[idx] + sigm(gi) * tanhf(gg);
      float h  = sigm(go) * tanhf(c);
      s_c[idx] = c; s_h[idx] = h; s_hb[idx] = f2bf(h);
    }
    __syncthreads();

    // stage 3: heads (color 48 tiles = 6/wave; type 2 tiles on waves 0,1)
    {
      v16bf ah[8];
#pragma unroll
      for (int kt = 0; kt < 8; ++kt) ah[kt] = load_a_frag(&s_hb[am * 256], kt * 32, lane);
#pragma unroll 1
      for (int j = 0; j < 6; ++j) {
        int n_t = wv * 6 + j;                            // scalar
        const __bf16* Bc = Wcolb + (long long)n_t * 4096;
        v8f acc = {};
#pragma unroll
        for (int kt = 0; kt < 8; ++kt) {
          v16bf b = *(const v16bf*)(Bc + (kt * 32 + lane) * 16);
          acc = wmma_bf16(ah[kt], b, acc);
        }
#pragma unroll
        for (int r = 0; r < 8; ++r)
          s_gates[(dm0 + r) * 768 + n_t * 16 + dn] = acc[r] + b_color[n_t * 16 + dn];
      }
      if (wv < 2) {                                      // scalar branch, EXEC stays full
        const __bf16* Bt = Wtypb + (long long)wv * 4096;
        v8f acc = {};
#pragma unroll
        for (int kt = 0; kt < 8; ++kt) {
          v16bf b = *(const v16bf*)(Bt + (kt * 32 + lane) * 16);
          acc = wmma_bf16(ah[kt], b, acc);
        }
#pragma unroll
        for (int r = 0; r < 8; ++r)
          s_typ[(dm0 + r) * 32 + wv * 16 + dn] = acc[r] + b_type[wv * 16 + dn];
      }
    }
    __syncthreads();

    // stage 4a: stream logits to global; partial argmax / partial pos dot
    {
      const float* s_logits = s_gates;
      for (int idx = tid; idx < 16 * 768; idx += 256) {
        int m = idx / 768, c = idx % 768;
        out[OFF_COLOR + ((long long)(b0 + m) * 128 + t) * 768 + c] = s_logits[idx];
      }
      for (int idx = tid; idx < 512; idx += 256) {
        int m = idx >> 5, jj = idx & 31;
        out[OFF_TYPE + ((long long)(b0 + m) * 128 + t) * 32 + jj] = s_typ[idx];
      }
      // pos partials: t = m*16 + jj*4 + part, 64-element sub-dot each
      {
        int m = tid >> 4, jj = (tid >> 2) & 3, part = tid & 3;
        const float* hr = &s_h[m * 256 + part * 64];
        const float* wp = &W_pos[(part * 64) * 4 + jj];
        float acc = 0.0f;
#pragma unroll 4
        for (int k = 0; k < 64; ++k) acc += hr[k] * wp[k * 4];
        s_posp[tid] = acc;
      }
      // color argmax partials: t = m*12 + ch*4 + part (t < 192)
      if (tid < 192) {
        int m = tid / 12, r = tid % 12, ch = r >> 2, part = r & 3;
        const float* p = &s_logits[m * 768 + ch * 256 + part * 64];
        int best = part * 64; float bv = p[0];
        for (int k = 1; k < 64; ++k) { float v = p[k]; if (v > bv) { bv = v; best = part * 64 + k; } }
        s_redv[tid] = bv; s_redi[tid] = best;
      }
      if (tid >= 192 && tid < 208) {        // type argmax (32 wide, keep serial)
        int m = tid - 192;
        const float* p = &s_typ[m * 32];
        int best = 0; float bv = p[0];
        for (int k = 1; k < 32; ++k) { float v = p[k]; if (v > bv) { bv = v; best = k; } }
        s_tidx[m] = best;
      }
    }
    __syncthreads();

    // stage 4b: combine partials (first-max-wins preserved: ascending part order, strict >)
    {
      if (tid < 48) {
        int m = tid / 3, ch = tid % 3, base = m * 12 + ch * 4;
        float bv = s_redv[base]; int bi = s_redi[base];
#pragma unroll
        for (int p = 1; p < 4; ++p) {
          float v = s_redv[base + p];
          if (v > bv) { bv = v; bi = s_redi[base + p]; }
        }
        s_cidx[m * 3 + ch] = bi;
      }
      if (tid >= 64 && tid < 128) {
        int q = tid - 64, m = q >> 2, jj = q & 3, base = m * 16 + jj * 4;
        float acc = b_pos[jj] + s_posp[base] + s_posp[base + 1] + s_posp[base + 2] + s_posp[base + 3];
        s_pos[m * 4 + jj] = acc;
        out[OFF_POS + ((long long)(b0 + m) * 128 + t) * 4 + jj] = acc;
      }
    }
    __syncthreads();

    // stage 5: e_next = emb_type[ti] + sum(emb_color[ci]) + pos@W_pos_in + b_pos_in
    {
      int d = tid;
      float wp0 = W_pos_in[d], wp1 = W_pos_in[256 + d],
            wp2 = W_pos_in[512 + d], wp3 = W_pos_in[768 + d];
      float bp = b_pos_in[d];
      for (int m = 0; m < 16; ++m) {
        float v = bp + emb_type[s_tidx[m] * 256 + d]
                + emb_color[s_cidx[m * 3 + 0] * 256 + d]
                + emb_color[s_cidx[m * 3 + 1] * 256 + d]
                + emb_color[s_cidx[m * 3 + 2] * 256 + d]
                + s_pos[m * 4 + 0] * wp0 + s_pos[m * 4 + 1] * wp1
                + s_pos[m * 4 + 2] * wp2 + s_pos[m * 4 + 3] * wp3;
        s_eb[m * 256 + d] = f2bf(v);
      }
    }
    __syncthreads();
  }
}

// ---------- launcher ----------

extern "C" void kernel_launch(void* const* d_in, const int* in_sizes, int n_in,
                              void* d_out, int out_size, void* d_ws, size_t ws_size,
                              hipStream_t stream) {
  const float* z        = (const float*)d_in[0];
  const float* bos      = (const float*)d_in[1];
  const float* emb_type = (const float*)d_in[2];
  const float* emb_color= (const float*)d_in[3];
  const float* W_pos_in = (const float*)d_in[4];
  const float* b_pos_in = (const float*)d_in[5];
  const float* Wx       = (const float*)d_in[6];
  const float* Wh       = (const float*)d_in[7];
  const float* b_lstm   = (const float*)d_in[8];
  const float* Wh0      = (const float*)d_in[9];
  const float* bh0      = (const float*)d_in[10];
  const float* Wc0      = (const float*)d_in[11];
  const float* bc0      = (const float*)d_in[12];
  const float* W_len    = (const float*)d_in[13];
  const float* b_len    = (const float*)d_in[14];
  const float* W_type   = (const float*)d_in[15];
  const float* b_type   = (const float*)d_in[16];
  const float* W_pos    = (const float*)d_in[17];
  const float* b_pos    = (const float*)d_in[18];
  const float* W_color  = (const float*)d_in[19];
  const float* b_color  = (const float*)d_in[20];

  // bf16 packed weights in workspace (~1.75 MB)
  unsigned short* ws   = (unsigned short*)d_ws;
  unsigned short* Wxb  = ws;                  // 256*1024
  unsigned short* Whb  = Wxb  + 262144;       // 256*1024
  unsigned short* Wcb  = Whb  + 262144;       // 256*768
  unsigned short* Wtb  = Wcb  + 196608;       // 256*32
  unsigned short* Wh0b = Wtb  + 8192;         // 256*256
  unsigned short* Wc0b = Wh0b + 65536;        // 256*256
  unsigned short* Wlb  = Wc0b + 65536;        // 256*128

  auto pack = [&](const float* W, unsigned short* dst, int K, int N) {
    int total = K * N;
    hipLaunchKernelGGL(pack_wmma_b, dim3((total + 255) / 256), dim3(256), 0, stream,
                       W, dst, K, N);
  };
  pack(Wx, Wxb, 256, 1024);
  pack(Wh, Whb, 256, 1024);
  pack(W_color, Wcb, 256, 768);
  pack(W_type, Wtb, 256, 32);
  pack(Wh0, Wh0b, 256, 256);
  pack(Wc0, Wc0b, 256, 256);
  pack(W_len, Wlb, 256, 128);

  hipLaunchKernelGGL(ar_decoder_fused, dim3(32), dim3(256), 0, stream,
                     z, bos, emb_type, emb_color, W_pos_in, b_pos_in, b_lstm, bh0, bc0,
                     b_len, b_type, W_pos, b_pos, b_color,
                     (const unsigned short*)Wxb, (const unsigned short*)Whb,
                     (const unsigned short*)Wcb, (const unsigned short*)Wtb,
                     (const unsigned short*)Wh0b, (const unsigned short*)Wc0b,
                     (const unsigned short*)Wlb, (float*)d_out);
}
